// MACEInteraction_66460323938667
// MI455X (gfx1250) — matrix operand order repaired and checked by reading, not compiled, not run on hardware
//
#include <hip/hip_runtime.h>
#include <hip/hip_bf16.h>

typedef __attribute__((ext_vector_type(16))) _Float16     v16h;
typedef __attribute__((ext_vector_type(8)))  float        v8f;
typedef __attribute__((ext_vector_type(4)))  float        v4f;
typedef __attribute__((ext_vector_type(4)))  unsigned int u32x4;

#define INV_SQRT2  0.70710678118654752f
#define INV_SQRT3  0.57735026918962576f
#define INV_SQRT_C 0.125f
#define WPB  4      // waves per block in edge kernel
#define HPAD 72     // padded row length (halves) to avoid LDS bank conflicts

__device__ __forceinline__ void wave_lds_fence() {
  __builtin_amdgcn_wave_barrier();
  asm volatile("s_wait_dscnt 0" ::: "memory");
  __builtin_amdgcn_wave_barrier();
}

// Fast silu: x * rcp(1+exp(-x)).  v_exp_f32 + v_add + v_rcp_f32 + fused mul/cvt.
__device__ __forceinline__ float silu_f(float x) {
  return x * __builtin_amdgcn_rcpf(1.0f + __expf(-x));
}

__device__ __forceinline__ v8f wmma_f16(v16h a, v16h b, v8f c) {
  return __builtin_amdgcn_wmma_f32_16x16x32_f16(false, a, false, b, (short)0, c,
                                                false, false);
}

// Load a 16-half B-fragment (contiguous 32B) as two b128 loads.
__device__ __forceinline__ v16h load_frag16(const _Float16* p) {
  union { v16h h; u32x4 u[2]; } f;
  f.u[0] = *(const u32x4*)(p);
  f.u[1] = *(const u32x4*)(p + 8);
  return f.h;
}

// ---------------------------------------------------------------------------
// Weight precompute: f16 copies, W2/W3 transposed to [n][k] for fragment loads
// ---------------------------------------------------------------------------
__global__ __launch_bounds__(256) void initw_kernel(
    const float* __restrict__ W1, const float* __restrict__ W2,
    const float* __restrict__ W3, _Float16* __restrict__ W1h,
    _Float16* __restrict__ W2T, _Float16* __restrict__ W3T)
{
  int i = blockIdx.x * 256 + threadIdx.x;
  if (i < 512) W1h[i] = (_Float16)W1[i];                 // [k][n], 8x64
  if (i < 4096) {                                         // W2T[n][k], 64x64
    int n = i >> 6, k = i & 63;
    W2T[i] = (_Float16)W2[k * 64 + n];
  }
  if (i < 16384) {                                        // W3T[n][k], 256x64
    int n = i >> 6, k = i & 63;
    W3T[i] = (_Float16)W3[k * 256 + n];
  }
}

__global__ void zero_kernel(float* __restrict__ p, int n) {
  int i = blockIdx.x * blockDim.x + threadIdx.x;
  int stride = gridDim.x * blockDim.x;
  for (; i < n; i += stride) p[i] = 0.0f;
}

// ---------------------------------------------------------------------------
// Pre linear: x0[n][c], x1[n][m][c] (m-major for coalesced edge gathers)
// ---------------------------------------------------------------------------
__global__ __launch_bounds__(256) void pre_kernel(
    const float* __restrict__ nf, const float* __restrict__ W0,
    const float* __restrict__ W1, float* __restrict__ x0ws,
    float* __restrict__ x1ws)
{
  __shared__ float w0[4096];
  __shared__ float w1[4096];
  const int tid = threadIdx.x;
  for (int i = tid; i < 4096; i += 256) { w0[i] = W0[i]; w1[i] = W1[i]; }
  __syncthreads();

  const int n = blockIdx.x;
  const int d = tid & 63;
  const int comp = tid >> 6;
  const float* row = nf + (size_t)n * 256;
  float acc = 0.0f;
  if (comp == 0) {
    #pragma unroll 8
    for (int k = 0; k < 64; ++k) acc += row[k] * w0[k * 64 + d];
    x0ws[(size_t)n * 64 + d] = acc * INV_SQRT_C;
  } else {
    const int m = comp - 1;
    #pragma unroll 8
    for (int k = 0; k < 64; ++k) acc += row[64 + 3 * k + m] * w1[k * 64 + d];
    x1ws[((size_t)n * 3 + m) * 64 + d] = acc * INV_SQRT_C;
  }
}

// ---------------------------------------------------------------------------
// Edge kernel: WMMA MLP (8->64->64->256) + message formation + atomic scatter
// One wave = one 16-edge tile per iteration.
// ---------------------------------------------------------------------------
__global__ __launch_bounds__(128) void edge_kernel(
    const float* __restrict__ rb, const float* __restrict__ sph,
    const int* __restrict__ eidx, const float* __restrict__ x0ws,
    const float* __restrict__ x1ws, const _Float16* __restrict__ W1h,
    const _Float16* __restrict__ W2T, const _Float16* __restrict__ W3T,
    float* __restrict__ agg0, float* __restrict__ agg1, int E)
{
  __shared__ __align__(16) _Float16 lds_w3[256 * HPAD];        // 36 KB
  __shared__ __align__(16) _Float16 lds_h[WPB][16 * HPAD];     // 4 x 2.25 KB

  const int tid = threadIdx.x;

  // Cooperative copy of W3T into LDS with padded rows (2048 chunks of 16B).
  #pragma unroll
  for (int q = 0; q < 16; ++q) {
    int ch = tid + q * 128;
    int n = ch >> 3, part = ch & 7;
    *(u32x4*)(&lds_w3[n * HPAD + part * 8]) =
        *(const u32x4*)(W3T + n * 64 + part * 8);
  }
  __syncthreads();

  const int lane = tid & 31;
  // Wave index is wave-uniform: force it scalar so the tile loop uses SALU
  // control flow instead of v_cmp + EXEC masking.
  const int wave = __builtin_amdgcn_readfirstlane(tid >> 5);
  const int hl = lane >> 4;     // half-wave: 0/1
  const int ln = lane & 15;

  // W1 as B-fragments, K padded 8->32 with zeros.  k = 16*hl + i.
  // All k >= 8 entries are exact 0.0 — this makes the A-side padding lanes
  // multiply to zero regardless of their (finite) contents.
  v16h b1[4];
  #pragma unroll
  for (int j = 0; j < 4; ++j) {
    #pragma unroll
    for (int i = 0; i < 16; ++i)
      b1[j][i] = (hl == 0 && i < 8) ? W1h[i * 64 + 16 * j + ln]
                                    : (_Float16)0.0f;
  }
  // W2 B-fragments resident in registers: 4 n-tiles x 2 k-tiles.
  v16h b2[4][2];
  #pragma unroll
  for (int j = 0; j < 4; ++j)
    #pragma unroll
    for (int t = 0; t < 2; ++t)
      b2[j][t] = load_frag16(W2T + (16 * j + ln) * 64 + 32 * t + 16 * hl);

  _Float16* hrow = &lds_h[wave][0];
  const int ntile = E >> 4;                // E is a multiple of 16
  const int tw = WPB * gridDim.x;

  v8f zero;
  #pragma unroll
  for (int i = 0; i < 8; ++i) zero[i] = 0.0f;

  for (int tile = wave + WPB * blockIdx.x; tile < ntile; tile += tw) {
    const int e0 = tile << 4;

    // ---- per-edge data first: these only depend on e0, so issuing them here
    // hides their latency behind the entire 3-stage WMMA MLP below.
    int er[8], es[8];
    v4f shv[8];
    #pragma unroll
    for (int v = 0; v < 8; ++v) {
      int e = e0 + v + 8 * hl;
      er[v] = eidx[e];          // recv
      es[v] = eidx[E + e];      // send
      shv[v] = *(const v4f*)(sph + (size_t)e * 4);
    }

    // ---- stage 1: h1 = silu(rb @ W1), A has K padded 8->32 ----
    // Unconditional load on ALL lanes: every element of a1 that maps to
    // k >= 8 hits a zero in b1, and every loaded value is a finite float,
    // so no predication (EXEC save/restore) is needed.
    v16h a1;
    {
      v4f r0 = *(const v4f*)(rb + (size_t)(e0 + ln) * 8);
      v4f r1 = *(const v4f*)(rb + (size_t)(e0 + ln) * 8 + 4);
      #pragma unroll
      for (int i = 0; i < 4; ++i) {
        a1[i]      = (_Float16)r0[i];
        a1[4 + i]  = (_Float16)r1[i];
        a1[8 + i]  = (_Float16)r0[i];   // k>=8 lanes: multiplied by 0 in B
        a1[12 + i] = (_Float16)r1[i];
      }
    }

    v8f c1[4];
    #pragma unroll
    for (int j = 0; j < 4; ++j) c1[j] = wmma_f16(a1, b1[j], zero);

    // silu + stage into LDS (row = edge-in-tile, col = feature)
    #pragma unroll
    for (int j = 0; j < 4; ++j)
      #pragma unroll
      for (int v = 0; v < 8; ++v)
        hrow[(v + 8 * hl) * HPAD + 16 * j + ln] = (_Float16)silu_f(c1[j][v]);
    wave_lds_fence();

    // reload as A-fragments: element i -> k = 32t + 16*hl + i
    v16h a2[2];
    #pragma unroll
    for (int t = 0; t < 2; ++t) {
      const _Float16* p = hrow + ln * HPAD + 32 * t + 16 * hl;
      union { v16h h; u32x4 u[2]; } f;
      f.u[0] = *(const u32x4*)(p);
      f.u[1] = *(const u32x4*)(p + 8);
      a2[t] = f.h;
    }

    // ---- stage 2: h2 = silu(h1 @ W2) ----
    v8f c2[4];
    #pragma unroll
    for (int j = 0; j < 4; ++j)
      c2[j] = wmma_f16(a2[1], b2[j][1], wmma_f16(a2[0], b2[j][0], zero));

    #pragma unroll
    for (int j = 0; j < 4; ++j)
      #pragma unroll
      for (int v = 0; v < 8; ++v)
        hrow[(v + 8 * hl) * HPAD + 16 * j + ln] = (_Float16)silu_f(c2[j][v]);
    wave_lds_fence();

    v16h a3[2];
    #pragma unroll
    for (int t = 0; t < 2; ++t) {
      const _Float16* p = hrow + ln * HPAD + 32 * t + 16 * hl;
      union { v16h h; u32x4 u[2]; } f;
      f.u[0] = *(const u32x4*)(p);
      f.u[1] = *(const u32x4*)(p + 8);
      a3[t] = f.h;
    }

    // ---- stage 3: w = h2 @ W3 fused with message math + scatter ----
    // C-layout: lane ln holds channel c = 16q + ln, VGPR v holds edge v+8*hl.
    #pragma unroll
    for (int q = 0; q < 4; ++q) {
      v8f cw[4];   // 0:w00  1:w110  2:w01  3:w10
      #pragma unroll
      for (int g = 0; g < 4; ++g) {
        const int j = q + 4 * g;
        v8f acc = zero;
        #pragma unroll
        for (int t = 0; t < 2; ++t) {
          v16h bf = load_frag16(&lds_w3[(16 * j + ln) * HPAD + 32 * t + 16 * hl]);
          acc = wmma_f16(a3[t], bf, acc);
        }
        cw[g] = acc;
      }
      const int c = 16 * q + ln;
      #pragma unroll
      for (int v = 0; v < 8; ++v) {
        const int esv = es[v];
        const float xj0 = x0ws[(size_t)esv * 64 + c];
        const float* xp = x1ws + (size_t)esv * 192 + c;
        const float x1a = xp[0], x1b = xp[64], x1c = xp[128];
        const float s0 = shv[v].x, s1 = shv[v].y, s2 = shv[v].z, s3 = shv[v].w;
        const float w00 = cw[0][v], w110 = cw[1][v];
        const float w01 = cw[2][v], w10 = cw[3][v];
        const float dot = x1a * s1 + x1b * s2 + x1c * s3;
        const float m0 = (w00 * xj0 * s0 + w110 * dot * INV_SQRT3) * INV_SQRT2;
        atomicAdd(agg0 + (size_t)er[v] * 64 + c, m0);
        float* a1p = agg1 + (size_t)er[v] * 192 + c;
        atomicAdd(a1p,       (w01 * xj0 * s1 + w10 * x1a * s0) * INV_SQRT2);
        atomicAdd(a1p + 64,  (w01 * xj0 * s2 + w10 * x1b * s0) * INV_SQRT2);
        atomicAdd(a1p + 128, (w01 * xj0 * s3 + w10 * x1c * s0) * INV_SQRT2);
      }
    }
  }
}

// ---------------------------------------------------------------------------
// Post linear: out[n][d][0] = y0, out[n][d][1+m] = y1[...][m]
// ---------------------------------------------------------------------------
__global__ __launch_bounds__(256) void post_kernel(
    const float* __restrict__ agg0, const float* __restrict__ agg1,
    const float* __restrict__ W0, const float* __restrict__ W1,
    float* __restrict__ out)
{
  __shared__ float w0[4096];
  __shared__ float w1[4096];
  const int tid = threadIdx.x;
  for (int i = tid; i < 4096; i += 256) { w0[i] = W0[i]; w1[i] = W1[i]; }
  __syncthreads();

  const int n = blockIdx.x;
  const int d = tid & 63;
  const int comp = tid >> 6;
  float acc = 0.0f;
  if (comp == 0) {
    const float* a = agg0 + (size_t)n * 64;
    #pragma unroll 8
    for (int c = 0; c < 64; ++c) acc += a[c] * w0[c * 64 + d];
  } else {
    const float* a = agg1 + ((size_t)n * 3 + (comp - 1)) * 64;
    #pragma unroll 8
    for (int c = 0; c < 64; ++c) acc += a[c] * w1[c * 64 + d];
  }
  out[(size_t)n * 256 + d * 4 + comp] = acc * INV_SQRT_C;
}

// ---------------------------------------------------------------------------
extern "C" void kernel_launch(void* const* d_in, const int* in_sizes, int n_in,
                              void* d_out, int out_size, void* d_ws, size_t ws_size,
                              hipStream_t stream)
{
  (void)n_in; (void)out_size; (void)ws_size;
  const float* nf     = (const float*)d_in[0];
  const float* sph    = (const float*)d_in[1];
  const float* rb     = (const float*)d_in[2];
  const float* preW0  = (const float*)d_in[3];
  const float* preW1  = (const float*)d_in[4];
  const float* mW1    = (const float*)d_in[5];
  const float* mW2    = (const float*)d_in[6];
  const float* mW3    = (const float*)d_in[7];
  const float* postW0 = (const float*)d_in[8];
  const float* postW1 = (const float*)d_in[9];
  const int*   eidx   = (const int*)d_in[10];

  const int N = in_sizes[0] / 256;   // 4*C = 256
  const int E = in_sizes[2] / 8;     // R = 8

  char* ws = (char*)d_ws;
  size_t off = 0;
  float* x0ws = (float*)(ws + off); off += (size_t)N * 64 * 4;
  float* x1ws = (float*)(ws + off); off += (size_t)N * 192 * 4;
  float* agg0 = (float*)(ws + off); off += (size_t)N * 64 * 4;
  float* agg1 = (float*)(ws + off); off += (size_t)N * 192 * 4;
  _Float16* W1h = (_Float16*)(ws + off); off += 1024;
  _Float16* W2T = (_Float16*)(ws + off); off += 8192;
  _Float16* W3T = (_Float16*)(ws + off); off += 32768;

  initw_kernel<<<64, 256, 0, stream>>>(mW1, mW2, mW3, W1h, W2T, W3T);
  zero_kernel<<<2048, 256, 0, stream>>>(agg0, N * 256);   // agg0+agg1 contiguous
  pre_kernel<<<N, 256, 0, stream>>>(nf, preW0, preW1, x0ws, x1ws);
  edge_kernel<<<1600, 128, 0, stream>>>(rb, sph, eidx, x0ws, x1ws,
                                        W1h, W2T, W3T, agg0, agg1, E);
  post_kernel<<<N, 256, 0, stream>>>(agg0, agg1, postW0, postW1, (float*)d_out);
}
